// Decoder_70832600645729
// MI455X (gfx1250) — matrix operand order repaired and checked by reading, not compiled, not run on hardware
//
#include <hip/hip_runtime.h>
#include <hip/hip_bf16.h>
#include <math.h>

typedef __bf16 bf16_t;
typedef __attribute__((ext_vector_type(16))) __bf16 v16bf;
typedef __attribute__((ext_vector_type(8)))  __bf16 v8bf;
typedef __attribute__((ext_vector_type(8)))  float  v8f;

#define DH 1024      // hidden size
#define DB 32        // batch
#define DT 64        // sequence length
#define DV 32000     // vocab
#define NKB (DH/32)  // 32 K-blocks of 32 for 16x16x32 WMMA

// --------- WMMA helpers (gfx1250, wave32) ---------

static __device__ __forceinline__ v8f wmma_bf16f32(v16bf a, v16bf b, v8f c) {
  // (neg_a, A, neg_b, B, c_mod, C, reuse_a, reuse_b)
  return __builtin_amdgcn_wmma_f32_16x16x32_bf16(false, a, false, b, (short)0, c,
                                                 false, false);
}

// A-fragment (16x32 bf16): src row-major [rows, DH]; rows [row_base, row_base+16).
// Lane layout (contiguous k, same as B; not ISA table order): lane = m (0..15) | khalf<<4;
//   elem i -> K = kb*32 + khalf*16 + i
static __device__ __forceinline__ v16bf load_a_frag(const bf16_t* __restrict__ base,
                                                    int row_base, int kb, int lane) {
  int m  = lane & 15;
  int kh = lane >> 4;
  const bf16_t* p = base + (size_t)(row_base + m) * DH + kb * 32 + kh * 16;
  v8bf lo = *(const v8bf*)(p);
  v8bf hi = *(const v8bf*)(p + 8);
  v16bf a;
#pragma unroll
  for (int i = 0; i < 8; ++i) { a[i] = lo[i]; a[i + 8] = hi[i]; }
  return a;
}

// B-fragment (32x16 bf16): weights row-major [N, DH]; B[k][n] = w[col_base+n][k].
// Lane layout: lane = n (0..15) | khalf<<4; elem i -> K = kb*32 + khalf*16 + i
static __device__ __forceinline__ v16bf load_b_frag(const bf16_t* __restrict__ w,
                                                    int col_base, int kb, int lane) {
  int n  = lane & 15;
  int kh = lane >> 4;
  const bf16_t* p = w + (size_t)(col_base + n) * DH + kb * 32 + kh * 16;
  return *(const v16bf*)p;
}

static __device__ __forceinline__ float sigmoid_f(float x) {
  return 1.0f / (1.0f + __expf(-x));
}

// --------- kernels ---------

__global__ void cvt_f32_bf16_kernel(const float* __restrict__ src,
                                    bf16_t* __restrict__ dst, long long n) {
  long long i = (long long)blockIdx.x * blockDim.x + threadIdx.x;
  long long stride = (long long)gridDim.x * blockDim.x;
  for (; i < n; i += stride) dst[i] = (bf16_t)src[i];
}

__global__ void init_hidden_kernel(const float* __restrict__ enc_h,
                                   float* __restrict__ h, bf16_t* __restrict__ hb) {
  int i = blockIdx.x * blockDim.x + threadIdx.x;
  if (i < DB * DH) { float v = enc_h[i]; h[i] = v; hb[i] = (bf16_t)v; }
}

// x = relu(emb[tok]); tok = SOS(=0) at t==0, else target[b, t-1]
__global__ void embed_kernel(const int* __restrict__ target,
                             const float* __restrict__ emb,
                             bf16_t* __restrict__ xb, int t) {
  int i = blockIdx.x * blockDim.x + threadIdx.x;
  if (i >= DB * DH) return;
  int b = i / DH, k = i % DH;
  int tok = (t == 0) ? 0 : target[b * DT + (t - 1)];
  float v = emb[(size_t)tok * DH + k];
  xb[i] = (bf16_t)(v > 0.0f ? v : 0.0f);
}

// One wave per (M-tile, 16-column N-tile) of H: 6 gate GEMM tiles via WMMA
// (ir,iz,in from x@W_ih^T; hr,hz,hn from h@W_hh^T), GRU nonlinearity fused
// in registers. 6 accumulators = 48 VGPRs -> no spills.
// grid = (DH/16)*2/4 = 32 blocks x 128 threads (4 waves).
__global__ __launch_bounds__(128, 1) void gru_step_kernel(
    const bf16_t* __restrict__ xb, const bf16_t* __restrict__ hb,
    const bf16_t* __restrict__ wih, const bf16_t* __restrict__ whh,
    const float* __restrict__ b_ih, const float* __restrict__ b_hh,
    float* __restrict__ h, bf16_t* __restrict__ hb_out) {
  int lane = threadIdx.x & 31;
  int wave = threadIdx.x >> 5;
  int gid  = blockIdx.x * 4 + wave;   // 0..127
  int jt   = gid >> 1;                // N-tile 0..63
  int mt   = gid & 1;                 // M-tile 0..1
  int jbase = jt * 16;
  int rowb  = mt * 16;

  const v8f vzero = {0.f, 0.f, 0.f, 0.f, 0.f, 0.f, 0.f, 0.f};
  v8f acc[6];                         // ir, iz, in, hr, hz, hn
#pragma unroll
  for (int g = 0; g < 6; ++g) acc[g] = vzero;

  for (int kb = 0; kb < NKB; ++kb) {
    v16bf ax = load_a_frag(xb, rowb, kb, lane);
    v16bf ah = load_a_frag(hb, rowb, kb, lane);
#pragma unroll
    for (int g = 0; g < 3; ++g) {
      v16bf bi = load_b_frag(wih, g * DH + jbase, kb, lane);
      v16bf bh = load_b_frag(whh, g * DH + jbase, kb, lane);
      acc[g]     = wmma_bf16f32(ax, bi, acc[g]);
      acc[3 + g] = wmma_bf16f32(ah, bh, acc[3 + g]);
    }
  }

  // C/D layout: elem r -> M = khalf*8 + r (within tile), N = lane&15
  int n  = lane & 15;
  int kh = lane >> 4;
  int j  = jbase + n;
  float bir = b_ih[j],          bhr = b_hh[j];
  float biz = b_ih[DH + j],     bhz = b_hh[DH + j];
  float bin = b_ih[2 * DH + j], bhn = b_hh[2 * DH + j];

#pragma unroll
  for (int r = 0; r < 8; ++r) {
    int b = rowb + kh * 8 + r;
    float gir = acc[0][r] + bir;
    float giz = acc[1][r] + biz;
    float gin = acc[2][r] + bin;
    float ghr = acc[3][r] + bhr;
    float ghz = acc[4][r] + bhz;
    float ghn = acc[5][r] + bhn;
    float rg = sigmoid_f(gir + ghr);
    float zg = sigmoid_f(giz + ghz);
    float ng = tanhf(gin + rg * ghn);
    float hold = h[(size_t)b * DH + j];
    float hnew = (1.0f - zg) * ng + zg * hold;
    h[(size_t)b * DH + j] = hnew;
    hb_out[(size_t)b * DH + j] = (bf16_t)hnew;
  }
}

// logits[b, t, v] = hb @ fc_w^T + fc_b. One wave per TWO 16-wide vocab tiles
// (A-fragments reused across both). grid = DV/128 = 250 blocks x 128 threads.
__global__ __launch_bounds__(128, 1) void fc_logits_kernel(
    const bf16_t* __restrict__ hb, const bf16_t* __restrict__ fcw,
    const float* __restrict__ fcb, float* __restrict__ out, int t) {
  int lane = threadIdx.x & 31;
  int wave = threadIdx.x >> 5;
  int nt = (blockIdx.x * 4 + wave) * 2;   // 0,2,..,1998
  int vbase0 = nt * 16;
  int vbase1 = vbase0 + 16;

  const v8f vzero = {0.f, 0.f, 0.f, 0.f, 0.f, 0.f, 0.f, 0.f};
  v8f acc00 = vzero, acc10 = vzero;   // M-tile 0/1 x N-tile 0
  v8f acc01 = vzero, acc11 = vzero;   // M-tile 0/1 x N-tile 1

  int n  = lane & 15;
  int kh = lane >> 4;
  const bf16_t* brow0 = fcw + (size_t)(vbase0 + n) * DH + kh * 16;
  const bf16_t* brow1 = fcw + (size_t)(vbase1 + n) * DH + kh * 16;

  for (int kb = 0; kb < NKB; ++kb) {
    v16bf a0 = load_a_frag(hb, 0,  kb, lane);
    v16bf a1 = load_a_frag(hb, 16, kb, lane);
    v16bf b0 = *(const v16bf*)(brow0 + kb * 32);
    v16bf b1 = *(const v16bf*)(brow1 + kb * 32);
    if (kb + 1 < NKB) {
      __builtin_prefetch(brow0 + (kb + 1) * 32, 0, 1);  // global_prefetch_b8
      __builtin_prefetch(brow1 + (kb + 1) * 32, 0, 1);
    }
    acc00 = wmma_bf16f32(a0, b0, acc00);
    acc10 = wmma_bf16f32(a1, b0, acc10);
    acc01 = wmma_bf16f32(a0, b1, acc01);
    acc11 = wmma_bf16f32(a1, b1, acc11);
  }

  float bias0 = fcb[vbase0 + n];
  float bias1 = fcb[vbase1 + n];
#pragma unroll
  for (int r = 0; r < 8; ++r) {
    int b0 = kh * 8 + r;
    size_t o0 = ((size_t)b0 * DT + t) * DV;
    size_t o1 = ((size_t)(b0 + 16) * DT + t) * DV;
    out[o0 + vbase0 + n] = acc00[r] + bias0;
    out[o1 + vbase0 + n] = acc10[r] + bias0;
    out[o0 + vbase1 + n] = acc01[r] + bias1;
    out[o1 + vbase1 + n] = acc11[r] + bias1;
  }
}

// In-place log_softmax over V for one (b, t) row per block.
__global__ void log_softmax_kernel(float* __restrict__ out, int t) {
  int b = blockIdx.x;
  float* row = out + ((size_t)b * DT + t) * DV;
  __shared__ float red[256];
  int tid = threadIdx.x;

  float m = -INFINITY;
  for (int i = tid; i < DV; i += 256) m = fmaxf(m, row[i]);
  red[tid] = m; __syncthreads();
  for (int s = 128; s > 0; s >>= 1) {
    if (tid < s) red[tid] = fmaxf(red[tid], red[tid + s]);
    __syncthreads();
  }
  m = red[0]; __syncthreads();

  float sum = 0.0f;
  for (int i = tid; i < DV; i += 256) sum += __expf(row[i] - m);
  red[tid] = sum; __syncthreads();
  for (int s = 128; s > 0; s >>= 1) {
    if (tid < s) red[tid] += red[tid + s];
    __syncthreads();
  }
  float lse = m + __logf(red[0]);

  for (int i = tid; i < DV; i += 256) row[i] -= lse;
}

__global__ void copy_hidden_kernel(const float* __restrict__ h,
                                   float* __restrict__ out_h) {
  int i = blockIdx.x * blockDim.x + threadIdx.x;
  if (i < DB * DH) out_h[i] = h[i];
}

// --------- launch ---------

extern "C" void kernel_launch(void* const* d_in, const int* in_sizes, int n_in,
                              void* d_out, int out_size, void* d_ws, size_t ws_size,
                              hipStream_t stream) {
  const float* enc_h  = (const float*)d_in[1];  // [1,B,H]
  const int*   target = (const int*)d_in[2];    // [B,T]
  const float* emb    = (const float*)d_in[3];  // [V,H]
  const float* W_ih   = (const float*)d_in[4];  // [3H,H]
  const float* W_hh   = (const float*)d_in[5];  // [3H,H]
  const float* b_ih   = (const float*)d_in[6];  // [3H]
  const float* b_hh   = (const float*)d_in[7];  // [3H]
  const float* fc_w   = (const float*)d_in[8];  // [V,H]
  const float* fc_b   = (const float*)d_in[9];  // [V]
  float* out = (float*)d_out;                   // [B,T,V] ++ [1,B,H]

  char* ws = (char*)d_ws;
  size_t off = 0;
  auto alloc = [&](size_t bytes) -> void* {
    off = (off + 255) & ~(size_t)255;
    void* p = ws + off;
    off += bytes;
    return p;
  };
  bf16_t* fcw_bf = (bf16_t*)alloc((size_t)DV * DH * 2);
  bf16_t* wih_bf = (bf16_t*)alloc((size_t)3 * DH * DH * 2);
  bf16_t* whh_bf = (bf16_t*)alloc((size_t)3 * DH * DH * 2);
  float*  h      = (float*) alloc((size_t)DB * DH * 4);
  bf16_t* hbA    = (bf16_t*)alloc((size_t)DB * DH * 2);
  bf16_t* hbB    = (bf16_t*)alloc((size_t)DB * DH * 2);
  bf16_t* xb     = (bf16_t*)alloc((size_t)DB * DH * 2);

  cvt_f32_bf16_kernel<<<2048, 256, 0, stream>>>(fc_w, fcw_bf, (long long)DV * DH);
  cvt_f32_bf16_kernel<<<1024, 256, 0, stream>>>(W_ih, wih_bf, (long long)3 * DH * DH);
  cvt_f32_bf16_kernel<<<1024, 256, 0, stream>>>(W_hh, whh_bf, (long long)3 * DH * DH);
  init_hidden_kernel<<<(DB * DH + 255) / 256, 256, 0, stream>>>(enc_h, h, hbA);

  for (int t = 0; t < DT; ++t) {
    bf16_t* cur = (t & 1) ? hbB : hbA;
    bf16_t* nxt = (t & 1) ? hbA : hbB;
    embed_kernel<<<(DB * DH + 255) / 256, 256, 0, stream>>>(target, emb, xb, t);
    gru_step_kernel<<<DH / 32, 128, 0, stream>>>(xb, cur, wih_bf, whh_bf,
                                                 b_ih, b_hh, h, nxt);
    fc_logits_kernel<<<DV / 128, 128, 0, stream>>>(nxt, fcw_bf, fc_b, out, t);
    log_softmax_kernel<<<DB, 256, 0, stream>>>(out, t);
  }

  copy_hidden_kernel<<<(DB * DH + 255) / 256, 256, 0, stream>>>(
      h, out + (size_t)DB * DT * DV);
}
